// AdaptiveSinkhornKD_17970143167379
// MI455X (gfx1250) — matrix operand order repaired and checked by reading, not compiled, not run on hardware
//
#include <hip/hip_runtime.h>

typedef unsigned short ushort_t;
typedef unsigned int uint_t;

typedef __attribute__((ext_vector_type(16))) __bf16        v16bf;
typedef __attribute__((ext_vector_type(8)))  float          v8f;
typedef __attribute__((ext_vector_type(8)))  uint_t         v8u;   // one 32B A/B fragment chunk

#define B_      1024
#define K_      100
#define KP      128     // padded K dimension (4 x 32 WMMA K-steps)
#define NP      112     // padded N dimension (7 x 16 tiles)
#define NT      7       // N tiles
#define MT      64      // M tiles (1024/16)
#define TEMP_   4.0f
#define LAMBDA_ 0.5f
#define EPS_    0.05f
#define INV_EPS 20.0f
#define MAXIT   50
#define LOGCLIP (-18.420680743952367f)   // log(1e-8)

// B-fragment table: NT*4 ksteps*32 lanes*16 halves = 14336 halves = 28672 B
#define BFRAG_HALVES (NT * 4 * 32 * 16)
// A-fragment table: MT*4*32*16 halves = 131072 halves = 262144 B
#define AFRAG_HALVES (MT * 4 * 32 * 16)

// ---------------- ws layout (bytes, 32B-aligned) ----------------
#define OFF_CRAW   0u          // 100*100 f32  (normalized C)
#define OFF_KB1    40064u      // Bfrag table for Kmat^T  (u-phase)
#define OFF_KB2    68736u      // Bfrag table for Kmat    (v-phase)
#define OFF_LMU    97408u      // 1024*100 f32 log_mu (teacher)
#define OFF_LNU    507008u     // 1024*100 f32 log_nu (student)
#define OFF_U      916608u     // 1024*100 f32
#define OFF_V      1326208u    // 1024*100 f32
#define OFF_UMAX   1735808u    // 1024 f32
#define OFF_VMAX   1739904u    // 1024 f32
#define OFF_EU     1744000u    // Afrag table (bf16) for Eu
#define OFF_EV     2006144u    // Afrag table (bf16) for Ev
#define OFF_S      2268288u    // 1024*112 f32 (GEMM output, reused)

__device__ __forceinline__ float wave_max(float v) {
  #pragma unroll
  for (int m = 16; m > 0; m >>= 1) v = fmaxf(v, __shfl_xor(v, m, 32));
  return v;
}
__device__ __forceinline__ float wave_sum(float v) {
  #pragma unroll
  for (int m = 16; m > 0; m >>= 1) v += __shfl_xor(v, m, 32);
  return v;
}
__device__ __forceinline__ ushort_t f2bf(float f) {
  uint_t u = __float_as_uint(f);
  uint_t r = (u + 0x7FFFu + ((u >> 16) & 1u)) >> 16;
  return (ushort_t)r;
}
// A-fragment k index for chunk element e (0..15), lane-half hi, kstep s
__device__ __forceinline__ int afrag_k(int s, int hi, int e) {
  return s * 32 + e + 8 * hi + ((e >= 8) ? 8 : 0);
}

// -------- zero the scalar accumulators (d_out is poisoned) --------
__global__ void init_scalars_kernel(float* out) {
  out[0] = 0.0f; out[1] = 0.0f; out[2] = 0.0f;
}

// -------- cost matrix + bf16 kernel matrices in WMMA B-fragment order --------
__global__ void cost_kernel(const float* __restrict__ A, float* __restrict__ Craw,
                            float* __restrict__ Cout, ushort_t* __restrict__ Kb1,
                            ushort_t* __restrict__ Kb2) {
  __shared__ float red[32];
  int tid  = threadIdx.x;
  int lane = tid & 31, wave = tid >> 5;
  float lmax = 0.0f;
  for (int idx = tid; idx < K_ * K_; idx += 1024) {
    int i = idx / K_, j = idx % K_;
    float c = 0.0f;
    if (i != j) {
      float s = 0.5f * (A[i * K_ + j] + A[j * K_ + i]);
      c = (s > 20.0f) ? s : log1pf(expf(s));      // stable softplus
    }
    Craw[idx] = c;
    lmax = fmaxf(lmax, c);
  }
  lmax = wave_max(lmax);
  if (lane == 0) red[wave] = lmax;
  __syncthreads();
  if (tid == 0) {
    float m = red[0];
    for (int w = 1; w < 32; ++w) m = fmaxf(m, red[w]);
    red[0] = m;
  }
  __syncthreads();
  float inv = 1.0f / (red[0] + 1e-8f);
  for (int idx = tid; idx < K_ * K_; idx += 1024) {
    float c = Craw[idx] * inv;
    Cout[idx] = c;
    Craw[idx] = c;
  }
  __syncthreads();
  // Bfrag layout: idx = ((nt*4+s)*32 + lane)*16 + e ; element maps to (k = s*32+8*hi+e+(e>=8?8:0), col = nt*16+(lane&15))
  for (int idx = tid; idx < BFRAG_HALVES; idx += 1024) {
    int e   = idx & 15;
    int ln  = (idx >> 4) & 31;
    int s   = (idx >> 9) & 3;
    int nt  = idx >> 11;
    int col = nt * 16 + (ln & 15);
    int hi  = ln >> 4;
    int k   = s * 32 + 8 * hi + e + ((e >= 8) ? 8 : 0);
    ushort_t h1 = 0, h2 = 0;
    if (k < K_ && col < K_) {
      h1 = f2bf(expf(-Craw[col * K_ + k] * INV_EPS));  // Kmat^T[k][col] = Kmat[col,k]
      h2 = f2bf(expf(-Craw[k * K_ + col] * INV_EPS));  // Kmat[k][col]
    }
    Kb1[idx] = h1;
    Kb2[idx] = h2;
  }
}

// -------- per-row softmaxes (temp 4) -> log_mu/log_nu, plus CE on raw student logits --------
__global__ void rowprep_kernel(const float* __restrict__ student, const float* __restrict__ teacher,
                               const int* __restrict__ labels, float* __restrict__ log_mu,
                               float* __restrict__ log_nu, float* __restrict__ out_scalars) {
  __shared__ float ce_part[8];
  int wave = threadIdx.x >> 5, lane = threadIdx.x & 31;
  int b = blockIdx.x * 8 + wave;

  float tv[4], sv[4], xv[4];
  #pragma unroll
  for (int t = 0; t < 4; ++t) {
    int i = lane + 32 * t;
    bool ok = i < K_;
    float xs = ok ? student[b * K_ + i] : 0.0f;
    float xt = ok ? teacher[b * K_ + i] : 0.0f;
    tv[t] = ok ? xt * (1.0f / TEMP_) : -1e30f;
    sv[t] = ok ? xs * (1.0f / TEMP_) : -1e30f;
    xv[t] = ok ? xs : -1e30f;
  }
  // teacher -> log_mu
  float m = wave_max(fmaxf(fmaxf(tv[0], tv[1]), fmaxf(tv[2], tv[3])));
  float s = 0.0f;
  #pragma unroll
  for (int t = 0; t < 4; ++t) if (lane + 32 * t < K_) s += expf(tv[t] - m);
  float lse = logf(wave_sum(s));
  #pragma unroll
  for (int t = 0; t < 4; ++t) {
    int i = lane + 32 * t;
    if (i < K_) log_mu[b * K_ + i] = fmaxf(tv[t] - m - lse, LOGCLIP);
  }
  // student -> log_nu
  m = wave_max(fmaxf(fmaxf(sv[0], sv[1]), fmaxf(sv[2], sv[3])));
  s = 0.0f;
  #pragma unroll
  for (int t = 0; t < 4; ++t) if (lane + 32 * t < K_) s += expf(sv[t] - m);
  lse = logf(wave_sum(s));
  #pragma unroll
  for (int t = 0; t < 4; ++t) {
    int i = lane + 32 * t;
    if (i < K_) log_nu[b * K_ + i] = fmaxf(sv[t] - m - lse, LOGCLIP);
  }
  // CE on raw logits
  float m2 = wave_max(fmaxf(fmaxf(xv[0], xv[1]), fmaxf(xv[2], xv[3])));
  s = 0.0f;
  #pragma unroll
  for (int t = 0; t < 4; ++t) if (lane + 32 * t < K_) s += expf(xv[t] - m2);
  float lse2 = logf(wave_sum(s));
  if (lane == 0) {
    int lbl = labels[b];
    float xl = student[b * K_ + lbl];
    ce_part[wave] = -(xl - m2 - lse2);
  }
  __syncthreads();
  if (threadIdx.x == 0) {
    float acc = 0.0f;
    for (int w = 0; w < 8; ++w) acc += ce_part[w];
    atomicAdd(out_scalars + 2, acc * (1.0f / (float)B_));
  }
}

// -------- Ev A-fragment table = exp((0-0)/eps)=1 for k<100, 0 pad; vmax = 0 --------
__global__ void init_ea_kernel(ushort_t* __restrict__ EA, float* __restrict__ vmax) {
  int idx = blockIdx.x * 256 + threadIdx.x;   // AFRAG_HALVES entries
  int e  = idx & 15;
  int ln = (idx >> 4) & 31;
  int s  = (idx >> 9) & 3;
  int hi = ln >> 4;
  int k  = afrag_k(s, hi, e);
  EA[idx] = (k < K_) ? (ushort_t)0x3F80 : (ushort_t)0;  // bf16(1.0)
  if (idx < B_) vmax[idx] = 0.0f;
}

// -------- bf16 WMMA GEMM: S(1024x112,f32) = E(Afrag) x Bmat(Bfrag) --------
__global__ void gemm_wmma_kernel(const v8u* __restrict__ Afrag,
                                 const uint4* __restrict__ Bfrag,
                                 float* __restrict__ Sout) {
  __attribute__((aligned(32))) __shared__ ushort_t Blds[BFRAG_HALVES];   // 28672 bytes
  // async DMA global -> LDS, 16B per lane per issue (ASYNCcnt-tracked)
  for (int c = threadIdx.x; c < BFRAG_HALVES / 8; c += 256) {   // 1792 chunks of 16B
    uint_t ldsaddr = (uint_t)(uintptr_t)(&Blds[c * 8]);
    const uint4* gsrc = Bfrag + c;
    asm volatile("global_load_async_to_lds_b128 %0, %1, off"
                 :: "v"(ldsaddr), "v"(gsrc) : "memory");
  }
  asm volatile("s_wait_asynccnt 0" ::: "memory");
  __syncthreads();

  int wave = threadIdx.x >> 5, lane = threadIdx.x & 31;
  int tile = blockIdx.x * 8 + wave;            // 448 tiles = 64 mtiles x 7 ntiles
  int mtile = tile / NT, ntile = tile % NT;
  int laneHi = lane >> 4, lm = lane & 15;
  int col = ntile * 16 + lm;                   // B/D column (N)

  v8f acc = {};
  #pragma unroll
  for (int s = 0; s < 4; ++s) {
    v8u a = Afrag[(mtile * 4 + s) * 32 + lane];                       // 2x global_load_b128
    v8u bfr = *(const v8u*)&Blds[((ntile * 4 + s) * 32 + lane) * 16]; // 2x ds_load_b128
    acc = __builtin_amdgcn_wmma_f32_16x16x32_bf16(
        false, __builtin_bit_cast(v16bf, a),
        false, __builtin_bit_cast(v16bf, bfr),
        (short)0, acc, false, false);
  }
  // D layout: VGPR v -> M = v + 8*laneHi, N = lane&15
  #pragma unroll
  for (int v = 0; v < 8; ++v) {
    int m = mtile * 16 + v + (laneHi ? 8 : 0);
    Sout[m * NP + col] = acc[v];
  }
}

// -------- Sinkhorn epilogue: uv = eps*logT - othermax - eps*log(S); emit exp in A-frag order --------
__global__ void update_kernel(const float* __restrict__ Sin, const float* __restrict__ logtarget,
                              const float* __restrict__ othermax, float* __restrict__ uv_out,
                              float* __restrict__ mymax_out, ushort_t* __restrict__ EA) {
  __shared__ ushort_t rowbuf[8][KP];   // per-wave staging of one row of exp-factors
  int wave = threadIdx.x >> 5, lane = threadIdx.x & 31;
  int b = blockIdx.x * 8 + wave;
  float om = othermax[b];
  float ui[4];
  float lmax = -1e30f;
  #pragma unroll
  for (int t = 0; t < 4; ++t) {
    int i = lane + 32 * t;
    if (i < K_) {
      float sv = Sin[b * NP + i];
      float x = EPS_ * logtarget[b * K_ + i] - om - EPS_ * logf(fmaxf(sv, 1e-38f));
      ui[t] = x;
      uv_out[b * K_ + i] = x;
      lmax = fmaxf(lmax, x);
    } else {
      ui[t] = -1e30f;
    }
  }
  float um = wave_max(lmax);
  if (lane == 0) mymax_out[b] = um;
  #pragma unroll
  for (int t = 0; t < 4; ++t) {
    int i = lane + 32 * t;
    float val = (i < K_) ? expf((ui[t] - um) * INV_EPS) : 0.0f;
    rowbuf[wave][i] = f2bf(val);
  }
  // lanes 0..7 pack this row's 8 A-fragment chunks (32B each); per-wave LDS is in-order
  if (lane < 8) {
    int s  = lane >> 1;
    int hi = lane & 1;
    int mtile = b >> 4, lm = b & 15;
    v8u pk;
    #pragma unroll
    for (int w = 0; w < 8; ++w) {
      uint_t lo = rowbuf[wave][afrag_k(s, hi, 2 * w)];
      uint_t hi16 = rowbuf[wave][afrag_k(s, hi, 2 * w + 1)];
      pk[w] = lo | (hi16 << 16);
    }
    *(v8u*)&EA[((mtile * 4 + s) * 32 + lm + 16 * hi) * 16] = pk;
  }
}

// -------- final plan pi = exp((-C + u + v)/eps), OT cost = mean_b sum_ij pi*C --------
__global__ void plan_kernel(const float* __restrict__ u, const float* __restrict__ v,
                            const float* __restrict__ C, float* __restrict__ plan,
                            float* __restrict__ out_scalars) {
  __shared__ float part[8];
  int wave = threadIdx.x >> 5, lane = threadIdx.x & 31;
  int gid = blockIdx.x * 8 + wave;   // (b,i) pair, 1024*100 total
  int b = gid / K_, i = gid % K_;
  float ub = u[b * K_ + i];
  float acc = 0.0f;
  #pragma unroll
  for (int t = 0; t < 4; ++t) {
    int j = lane + 32 * t;
    if (j < K_) {
      float c  = C[i * K_ + j];
      float pij = expf((-c + ub + v[b * K_ + j]) * INV_EPS);
      plan[(size_t)gid * K_ + j] = pij;
      acc += pij * c;
    }
  }
  acc = wave_sum(acc);
  if (lane == 0) part[wave] = acc;
  __syncthreads();
  if (threadIdx.x == 0) {
    float s = 0.0f;
    for (int w = 0; w < 8; ++w) s += part[w];
    atomicAdd(out_scalars + 1, s * (1.0f / (float)B_));
  }
}

__global__ void finalize_kernel(float* out) {
  out[0] = out[2] + LAMBDA_ * out[1];
}

extern "C" void kernel_launch(void* const* d_in, const int* in_sizes, int n_in,
                              void* d_out, int out_size, void* d_ws, size_t ws_size,
                              hipStream_t stream) {
  (void)in_sizes; (void)n_in; (void)out_size; (void)ws_size;
  const float* student = (const float*)d_in[0];
  const float* teacher = (const float*)d_in[1];
  const float* A       = (const float*)d_in[2];
  const int*   labels  = (const int*)d_in[3];

  float* out = (float*)d_out;
  float* plan_out = out + 3;                               // 1024*100*100
  float* C_out    = out + 3 + (size_t)B_ * K_ * K_;        // 100*100

  char* ws = (char*)d_ws;
  float*    Craw = (float*)(ws + OFF_CRAW);
  ushort_t* Kb1  = (ushort_t*)(ws + OFF_KB1);
  ushort_t* Kb2  = (ushort_t*)(ws + OFF_KB2);
  float*    lmu  = (float*)(ws + OFF_LMU);
  float*    lnu  = (float*)(ws + OFF_LNU);
  float*    u    = (float*)(ws + OFF_U);
  float*    v    = (float*)(ws + OFF_V);
  float*    umax = (float*)(ws + OFF_UMAX);
  float*    vmax = (float*)(ws + OFF_VMAX);
  ushort_t* Eu   = (ushort_t*)(ws + OFF_EU);
  ushort_t* Ev   = (ushort_t*)(ws + OFF_EV);
  float*    S    = (float*)(ws + OFF_S);

  init_scalars_kernel<<<1, 1, 0, stream>>>(out);
  cost_kernel<<<1, 1024, 0, stream>>>(A, Craw, C_out, Kb1, Kb2);
  rowprep_kernel<<<B_ / 8, 256, 0, stream>>>(student, teacher, labels, lmu, lnu, out);
  init_ea_kernel<<<AFRAG_HALVES / 256, 256, 0, stream>>>(Ev, vmax);

  for (int it = 0; it < MAXIT; ++it) {
    // u-update: S = Ev @ Kmat^T, then u = eps*log_mu - vmax - eps*log S
    gemm_wmma_kernel<<<56, 256, 0, stream>>>((const v8u*)Ev, (const uint4*)Kb1, S);
    update_kernel<<<B_ / 8, 256, 0, stream>>>(S, lmu, vmax, u, umax, Eu);
    // v-update: T = Eu @ Kmat, then v = eps*log_nu - umax - eps*log T
    gemm_wmma_kernel<<<56, 256, 0, stream>>>((const v8u*)Eu, (const uint4*)Kb2, S);
    update_kernel<<<B_ / 8, 256, 0, stream>>>(S, lnu, umax, v, vmax, Ev);
  }

  plan_kernel<<<(B_ * K_) / 8, 256, 0, stream>>>(u, v, Craw, plan_out, out);
  finalize_kernel<<<1, 1, 0, stream>>>(out);
}
